// SigmoidAttentionLayer_18287970746895
// MI455X (gfx1250) — matrix operand order repaired and checked by reading, not compiled, not run on hardware
//
#include <hip/hip_runtime.h>
#include <hip/hip_bf16.h>
#include <math.h>

// ---------------------------------------------------------------------------
// SigmoidAttention transformer layer for MI455X (gfx1250), wave32 + WMMA bf16.
//
// Roofline: ~137 GFLOP of GEMMs vs a mandatory 512MB f32 attn_probs store
// (~22us at 23.3 TB/s HBM). All contractions run on v_wmma_f32_16x16x32_bf16
// (f32 accumulate). Attention is fused flash-style (sigmoid needs no row
// normalization). GEMMs are software-pipelined: GLOBAL_LOAD_ASYNC_TO_LDS_B128
// double-buffers LDS tiles, waiting only on asynccnt<=4 so stage t+1's
// global traffic overlaps stage t's WMMAs.
// ---------------------------------------------------------------------------

typedef __attribute__((ext_vector_type(16))) __bf16 v16bf;
typedef __attribute__((ext_vector_type(8)))  __bf16 v8bf;
typedef __attribute__((ext_vector_type(8)))  float  v8f;

#define D_MODEL   1024
#define N_SEQ     2048
#define BATCH     2
#define NHEAD     16
#define HEAD_DIM  64
#define D_FF      4096
#define ROWS      (BATCH * N_SEQ)          // 4096

#if defined(__has_builtin)
#  if __has_builtin(__builtin_amdgcn_global_load_async_to_lds_b128) && \
      __has_builtin(__builtin_amdgcn_s_wait_asynccnt)
#    define USE_ASYNC 1
#  endif
#endif
#ifndef USE_ASYNC
#  define USE_ASYNC 0
#endif

#if USE_ASYNC
// builtin signature (probe-confirmed via diagnostics): (int4 AS1*, int4 AS3*,
// imm offset, imm cpol); imm offset applies to BOTH global and LDS address.
typedef int v4i_gcc __attribute__((vector_size(16)));
typedef __attribute__((address_space(1))) v4i_gcc* as1_v4i;
typedef __attribute__((address_space(3))) v4i_gcc* as3_v4i;
#define AS1(p) ((as1_v4i)(p))
#define AS3(p) ((as3_v4i)(p))
#define ASYNC_CP64(gsrc, ldst)                                                   \
  do {                                                                           \
    __builtin_amdgcn_global_load_async_to_lds_b128(AS1(gsrc), AS3(ldst), 0, 0);  \
    __builtin_amdgcn_global_load_async_to_lds_b128(AS1(gsrc), AS3(ldst), 16, 0); \
    __builtin_amdgcn_global_load_async_to_lds_b128(AS1(gsrc), AS3(ldst), 32, 0); \
    __builtin_amdgcn_global_load_async_to_lds_b128(AS1(gsrc), AS3(ldst), 48, 0); \
  } while (0)
#define ASYNC_CP32(gsrc, ldst)                                                   \
  do {                                                                           \
    __builtin_amdgcn_global_load_async_to_lds_b128(AS1(gsrc), AS3(ldst), 0, 0);  \
    __builtin_amdgcn_global_load_async_to_lds_b128(AS1(gsrc), AS3(ldst), 16, 0); \
  } while (0)
#define ASYNC_WAIT_N(n) __builtin_amdgcn_s_wait_asynccnt(n)
#else
#define ASYNC_CP64(gsrc, ldst)                                              \
  do {                                                                      \
    const v8bf* _s = (const v8bf*)(gsrc);                                   \
    v8bf* _d = (v8bf*)(ldst);                                               \
    _d[0] = _s[0]; _d[1] = _s[1]; _d[2] = _s[2]; _d[3] = _s[3];             \
  } while (0)
#define ASYNC_CP32(gsrc, ldst)                                              \
  do {                                                                      \
    const v8bf* _s = (const v8bf*)(gsrc);                                   \
    v8bf* _d = (v8bf*)(ldst);                                               \
    _d[0] = _s[0]; _d[1] = _s[1];                                           \
  } while (0)
#define ASYNC_WAIT_N(n) do {} while (0)
#endif

__device__ __forceinline__ v16bf mk16(v8bf lo, v8bf hi) {
  return __builtin_shufflevector(lo, hi, 0,1,2,3,4,5,6,7,8,9,10,11,12,13,14,15);
}

__device__ __forceinline__ v8f wmma_bf16(v16bf a, v16bf b, v8f c) {
  return __builtin_amdgcn_wmma_f32_16x16x32_bf16(false, a, false, b, (short)0, c,
                                                 false, false);
}

__device__ __forceinline__ float fast_sigmoid(float x) {
  return __builtin_amdgcn_rcpf(1.0f + __expf(-x));   // v_exp + v_rcp
}

// --------------------- f32 -> bf16 cast with transpose ---------------------
__global__ __launch_bounds__(256)
void k_cast_t_bf16(const float* __restrict__ in, __bf16* __restrict__ out,
                   int R, int C) {
  __shared__ float tile[32][33];
  const int c0 = blockIdx.x * 32, r0 = blockIdx.y * 32;
  const int tx = threadIdx.x & 31, ty = threadIdx.x >> 5;  // 32 x 8
#pragma unroll
  for (int i = 0; i < 32; i += 8)
    tile[ty + i][tx] = in[(size_t)(r0 + ty + i) * C + c0 + tx];
  __syncthreads();
#pragma unroll
  for (int i = 0; i < 32; i += 8)
    out[(size_t)(c0 + ty + i) * R + r0 + tx] = (__bf16)tile[tx][ty + i];
}

// --------------------------- LayerNorm + bf16 cast -------------------------
__global__ __launch_bounds__(256)
void k_layernorm_bf16(const float* __restrict__ x, const float* __restrict__ g,
                      const float* __restrict__ b, __bf16* __restrict__ out) {
  __shared__ float red[256];
  const int row = blockIdx.x;
  const int t = threadIdx.x;
  const float4 v = ((const float4*)(x + (size_t)row * D_MODEL))[t];

  red[t] = v.x + v.y + v.z + v.w;
  __syncthreads();
  for (int o = 128; o > 0; o >>= 1) {
    if (t < o) red[t] += red[t + o];
    __syncthreads();
  }
  const float mu = red[0] * (1.0f / (float)D_MODEL);
  __syncthreads();

  const float d0 = v.x - mu, d1 = v.y - mu, d2 = v.z - mu, d3 = v.w - mu;
  red[t] = d0 * d0 + d1 * d1 + d2 * d2 + d3 * d3;
  __syncthreads();
  for (int o = 128; o > 0; o >>= 1) {
    if (t < o) red[t] += red[t + o];
    __syncthreads();
  }
  const float rs = rsqrtf(red[0] * (1.0f / (float)D_MODEL) + 1e-5f);

  const float4 gg = ((const float4*)g)[t];
  const float4 bb = ((const float4*)b)[t];
  __bf16* o = out + (size_t)row * D_MODEL + t * 4;
  o[0] = (__bf16)(d0 * rs * gg.x + bb.x);
  o[1] = (__bf16)(d1 * rs * gg.y + bb.y);
  o[2] = (__bf16)(d2 * rs * gg.z + bb.z);
  o[3] = (__bf16)(d3 * rs * gg.w + bb.w);
}

// ------------------------------ bf16 WMMA GEMM -----------------------------
// C[M,N] = A[M,K] @ Bt[N][K]^T + bias (+GELU) (+residual).
// 128x128 tile, K-step 32, double-buffered async LDS staging (40KB LDS).
// Wave grid 2x4: each wave owns 64x32 = 4x2 WMMA tiles, 8 WMMAs per stage.
// Pipeline: issue stage t+1 (4 async b128/thread), s_wait_asynccnt 4 keeps
// them in flight while stage t's data is guaranteed resident (in-order
// completion), barrier, compute.
template <int ACT, int RES, int OBF>
__global__ __launch_bounds__(256)
void k_gemm_bf16(const __bf16* __restrict__ A, const __bf16* __restrict__ Bt,
                 const float* __restrict__ bias,
                 const float* __restrict__ resid,
                 float* __restrict__ Cf, __bf16* __restrict__ Cb,
                 int M, int Nn, int K) {
  __shared__ __bf16 As[2][128][40];   // stride 40: 16B-aligned, conflict-free
  __shared__ __bf16 Bs[2][128][40];

  const int tid  = threadIdx.x;
  const int lane = tid & 31;
  const int w    = tid >> 5;
  const int wm   = w & 1;          // 64-row block
  const int wn   = w >> 1;         // 32-col block
  const int m0   = blockIdx.y * 128;
  const int n0   = blockIdx.x * 128;
  const int kh   = lane >> 4;
  const int lr   = lane & 15;

  const int ldrow = tid >> 1, ldk = (tid & 1) * 16;  // 32B per thread/array

  const __bf16* aRow = A  + (size_t)(m0 + ldrow) * K + ldk;
  const __bf16* bRow = Bt + (size_t)(n0 + ldrow) * K + ldk;

  v8f acc[4][2] = {};

  // prologue: stage 0 into buffer 0
  ASYNC_CP32(aRow, &As[0][ldrow][ldk]);
  ASYNC_CP32(bRow, &Bs[0][ldrow][ldk]);

  const int T = K >> 5;
  for (int t = 0; t < T; ++t) {
    const int cur = t & 1;
    if (t + 1 < T) {  // issue next stage, keep its 4 loads in flight
      const int kn = (t + 1) << 5;
      ASYNC_CP32(aRow + kn, &As[cur ^ 1][ldrow][ldk]);
      ASYNC_CP32(bRow + kn, &Bs[cur ^ 1][ldrow][ldk]);
      ASYNC_WAIT_N(4);
    } else {
      ASYNC_WAIT_N(0);
    }
    __syncthreads();

    v16bf afrag[4], bfrag[2];
#pragma unroll
    for (int i = 0; i < 4; ++i) {
      const int r = wm * 64 + i * 16 + lr;
      afrag[i] = mk16(*(const v8bf*)&As[cur][r][kh * 8],
                      *(const v8bf*)&As[cur][r][16 + kh * 8]);
    }
#pragma unroll
    for (int j = 0; j < 2; ++j) {
      const int c = wn * 32 + j * 16 + lr;
      bfrag[j] = mk16(*(const v8bf*)&Bs[cur][c][kh * 8],
                      *(const v8bf*)&Bs[cur][c][16 + kh * 8]);
    }
#pragma unroll
    for (int i = 0; i < 4; ++i)
#pragma unroll
      for (int j = 0; j < 2; ++j)
        acc[i][j] = wmma_bf16(afrag[i], bfrag[j], acc[i][j]);
    __syncthreads();
  }

  // epilogue: C layout — vgpr jj holds row (kh*8 + jj) of the 16x16 tile
#pragma unroll
  for (int i = 0; i < 4; ++i)
#pragma unroll
    for (int j = 0; j < 2; ++j) {
      const int col  = n0 + wn * 32 + j * 16 + lr;
      const int row0 = m0 + wm * 64 + i * 16 + kh * 8;
      const float bi = bias[col];
      if (OBF == 2) {
        v8bf ob;
#pragma unroll
        for (int jj = 0; jj < 8; ++jj) ob[jj] = (__bf16)(acc[i][j][jj] + bi);
        *(v8bf*)&Cb[(size_t)col * M + row0] = ob;   // transposed, contiguous
      } else {
#pragma unroll
        for (int jj = 0; jj < 8; ++jj) {
          const int row = row0 + jj;
          float val = acc[i][j][jj] + bi;
          if (ACT == 1) val = 0.5f * val * (1.0f + erff(val * 0.70710678118f));
          if (RES) val += resid[(size_t)row * Nn + col];
          if (OBF == 1) Cb[(size_t)row * Nn + col] = (__bf16)val;
          else          Cf[(size_t)row * Nn + col] = val;
        }
      }
    }
}

// -------------------------- fused sigmoid attention ------------------------
// grid.x = b*h (32), grid.y = q-tile of 64 (32). 8 waves per block.
// vt is the V projection stored transposed: vt[col][row], col = h*64+d.
__global__ __launch_bounds__(256)
void k_attn(const __bf16* __restrict__ q, const __bf16* __restrict__ k,
            const __bf16* __restrict__ vt, float* __restrict__ probs,
            __bf16* __restrict__ ao) {
  __shared__ __bf16 Qs[64][72];     // [q][d]
  __shared__ __bf16 Ks[128][72];    // [key][d]  (natural B layout for Q@K^T)
  __shared__ __bf16 Vt[64][136];    // [d][key]  (natural B layout for P@V)
  __shared__ __bf16 Ps[64][136];    // [q][key]  bf16 probs tile

  const int tid  = threadIdx.x;
  const int lane = tid & 31;
  const int w    = tid >> 5;
  const int lr   = lane & 15;
  const int kh   = lane >> 4;

  const int bh = blockIdx.x;                  // 0..31
  const int b  = bh >> 4, h = bh & 15;
  const int q0 = blockIdx.y * 64;
  const int rowbase = b * N_SEQ;
  const int cbase   = h * HEAD_DIM;

  // stage Q tile once (async)
  {
    const int r = tid >> 2, c = (tid & 3) * 16;
    ASYNC_CP32(q + (size_t)(rowbase + q0 + r) * D_MODEL + cbase + c, &Qs[r][c]);
  }

  const int sm = w & 1, sn = w >> 1;    // S wave grid: 2x4 of 32x32
  const int om = w & 3, on = w >> 2;    // O wave grid: 4x2 of 16x32
  v8f oacc[2] = {};

  for (int kt = 0; kt < 16; ++kt) {
    const int kr0 = kt * 128;
    {  // K tile 128x64 (row-major per key)
      const int r = tid >> 1, c = (tid & 1) * 32;
      ASYNC_CP64(k + (size_t)(rowbase + kr0 + r) * D_MODEL + cbase + c,
                 &Ks[r][c]);
    }
    {  // V tile already transposed in memory: Vt[d][key]
      const int d = tid >> 2, c = (tid & 3) * 32;
      ASYNC_CP64(vt + (size_t)(cbase + d) * ROWS + rowbase + kr0 + c,
                 &Vt[d][c]);
    }
    ASYNC_WAIT_N(0);
    __syncthreads();

    // S = Q @ K^T : each wave 2x2 WMMA tiles, contraction over d = 64
    v8f s[2][2] = {};
#pragma unroll
    for (int ks = 0; ks < 64; ks += 32) {
      v16bf aq[2], bk2[2];
#pragma unroll
      for (int i = 0; i < 2; ++i) {
        const int r = sm * 32 + i * 16 + lr;
        aq[i] = mk16(*(const v8bf*)&Qs[r][ks + kh * 8],
                     *(const v8bf*)&Qs[r][ks + 16 + kh * 8]);
      }
#pragma unroll
      for (int j = 0; j < 2; ++j) {
        const int c = sn * 32 + j * 16 + lr;
        bk2[j] = mk16(*(const v8bf*)&Ks[c][ks + kh * 8],
                      *(const v8bf*)&Ks[c][ks + 16 + kh * 8]);
      }
#pragma unroll
      for (int i = 0; i < 2; ++i)
#pragma unroll
        for (int j = 0; j < 2; ++j)
          s[i][j] = wmma_bf16(aq[i], bk2[j], s[i][j]);
    }

    // sigmoid (v_exp + v_rcp), stream f32 probs to HBM, stash bf16 in LDS
#pragma unroll
    for (int i = 0; i < 2; ++i)
#pragma unroll
      for (int j = 0; j < 2; ++j) {
        const int kc = sn * 32 + j * 16 + lr;
#pragma unroll
        for (int jj = 0; jj < 8; ++jj) {
          const int qr = sm * 32 + i * 16 + kh * 8 + jj;
          const float val = fast_sigmoid(s[i][j][jj] * 0.125f);  // 1/sqrt(64)
          probs[((size_t)bh * N_SEQ + q0 + qr) * N_SEQ + kr0 + kc] = val;
          Ps[qr][kc] = (__bf16)val;
        }
      }
    __syncthreads();

    // O += P @ V : each wave 1x2 tiles, contraction over 128 keys
#pragma unroll
    for (int ks = 0; ks < 128; ks += 32) {
      const int r = om * 16 + lr;
      v16bf ap = mk16(*(const v8bf*)&Ps[r][ks + kh * 8],
                      *(const v8bf*)&Ps[r][ks + 16 + kh * 8]);
#pragma unroll
      for (int j = 0; j < 2; ++j) {
        const int c = on * 32 + j * 16 + lr;
        v16bf bv = mk16(*(const v8bf*)&Vt[c][ks + kh * 8],
                        *(const v8bf*)&Vt[c][ks + 16 + kh * 8]);
        oacc[j] = wmma_bf16(ap, bv, oacc[j]);
      }
    }
    __syncthreads();
  }

  // write attn_out (bf16, row = token, col = h*64+d) with 1/sqrt(N) scale
  const float inv_sqrt_n = 0.022097086912079608f;  // 1/sqrt(2048)
#pragma unroll
  for (int j = 0; j < 2; ++j) {
    const int col = cbase + on * 32 + j * 16 + lr;
#pragma unroll
    for (int jj = 0; jj < 8; ++jj) {
      const int row = rowbase + q0 + om * 16 + kh * 8 + jj;
      ao[(size_t)row * D_MODEL + col] = (__bf16)(oacc[j][jj] * inv_sqrt_n);
    }
  }
}

// ------------------------------- launcher ----------------------------------
extern "C" void kernel_launch(void* const* d_in, const int* in_sizes, int n_in,
                              void* d_out, int out_size, void* d_ws,
                              size_t ws_size, hipStream_t stream) {
  const float* x     = (const float*)d_in[0];
  const float* wq    = (const float*)d_in[1];
  const float* bq    = (const float*)d_in[2];
  const float* wk    = (const float*)d_in[3];
  const float* bk    = (const float*)d_in[4];
  const float* wv    = (const float*)d_in[5];
  const float* bv    = (const float*)d_in[6];
  const float* wo    = (const float*)d_in[7];
  const float* bo    = (const float*)d_in[8];
  const float* ln1_g = (const float*)d_in[9];
  const float* ln1_b = (const float*)d_in[10];
  const float* ln2_g = (const float*)d_in[11];
  const float* ln2_b = (const float*)d_in[12];
  const float* w1    = (const float*)d_in[13];
  const float* b1    = (const float*)d_in[14];
  const float* w2    = (const float*)d_in[15];
  const float* b2    = (const float*)d_in[16];

  float* out_x = (float*)d_out;                           // [2,2048,1024]
  float* probs = (float*)d_out + (size_t)ROWS * D_MODEL;  // [2,16,2048,2048]

  char* ws = (char*)d_ws;
  __bf16* wq_bf = (__bf16*)(ws + (size_t)0);               // all transposed
  __bf16* wk_bf = (__bf16*)(ws + (size_t)2  * 1024 * 1024);
  __bf16* wv_bf = (__bf16*)(ws + (size_t)4  * 1024 * 1024);
  __bf16* wo_bf = (__bf16*)(ws + (size_t)6  * 1024 * 1024);
  __bf16* w1_bf = (__bf16*)(ws + (size_t)8  * 1024 * 1024);
  __bf16* w2_bf = (__bf16*)(ws + (size_t)16 * 1024 * 1024);
  __bf16* nrm_bf = (__bf16*)(ws + (size_t)24 * 1024 * 1024);  // reused for LN2
  __bf16* q_bf  = (__bf16*)(ws + (size_t)32 * 1024 * 1024);
  __bf16* k_bf  = (__bf16*)(ws + (size_t)40 * 1024 * 1024);
  __bf16* vt_bf = (__bf16*)(ws + (size_t)48 * 1024 * 1024);   // transposed V
  __bf16* ao_bf = (__bf16*)(ws + (size_t)56 * 1024 * 1024);
  __bf16* h_bf  = (__bf16*)(ws + (size_t)64 * 1024 * 1024);
  float*  x1    = (float*)(ws + (size_t)96 * 1024 * 1024);

  // 1) weights -> bf16, transposed (Wt[n][k]); grid = (C/32, R/32)
  k_cast_t_bf16<<<dim3(32, 32),  256, 0, stream>>>(wq, wq_bf, 1024, 1024);
  k_cast_t_bf16<<<dim3(32, 32),  256, 0, stream>>>(wk, wk_bf, 1024, 1024);
  k_cast_t_bf16<<<dim3(32, 32),  256, 0, stream>>>(wv, wv_bf, 1024, 1024);
  k_cast_t_bf16<<<dim3(32, 32),  256, 0, stream>>>(wo, wo_bf, 1024, 1024);
  k_cast_t_bf16<<<dim3(128, 32), 256, 0, stream>>>(w1, w1_bf, 1024, 4096);
  k_cast_t_bf16<<<dim3(32, 128), 256, 0, stream>>>(w2, w2_bf, 4096, 1024);

  // 2) LN1
  k_layernorm_bf16<<<ROWS, 256, 0, stream>>>(x, ln1_g, ln1_b, nrm_bf);

  // 3) Q,K projections (bf16), V projection (bf16 transposed)
  dim3 gqkv(D_MODEL / 128, ROWS / 128);  // (8, 32)
  k_gemm_bf16<0, 0, 1><<<gqkv, 256, 0, stream>>>(nrm_bf, wq_bf, bq, nullptr,
      nullptr, q_bf, ROWS, D_MODEL, D_MODEL);
  k_gemm_bf16<0, 0, 1><<<gqkv, 256, 0, stream>>>(nrm_bf, wk_bf, bk, nullptr,
      nullptr, k_bf, ROWS, D_MODEL, D_MODEL);
  k_gemm_bf16<0, 0, 2><<<gqkv, 256, 0, stream>>>(nrm_bf, wv_bf, bv, nullptr,
      nullptr, vt_bf, ROWS, D_MODEL, D_MODEL);

  // 4) fused sigmoid attention (streams 512MB probs + produces attn_out)
  dim3 gattn(BATCH * NHEAD, N_SEQ / 64);  // (32, 32)
  k_attn<<<gattn, 256, 0, stream>>>(q_bf, k_bf, vt_bf, probs, ao_bf);

  // 5) output projection + residual -> x1 (f32)
  k_gemm_bf16<0, 1, 0><<<gqkv, 256, 0, stream>>>(ao_bf, wo_bf, bo, x,
      x1, nullptr, ROWS, D_MODEL, D_MODEL);

  // 6) LN2 (reuse nrm_bf)
  k_layernorm_bf16<<<ROWS, 256, 0, stream>>>(x1, ln2_g, ln2_b, nrm_bf);

  // 7) FFN up + exact GELU (bf16 out)
  dim3 gff1(D_FF / 128, ROWS / 128);  // (32, 32)
  k_gemm_bf16<1, 0, 1><<<gff1, 256, 0, stream>>>(nrm_bf, w1_bf, b1, nullptr,
      nullptr, h_bf, ROWS, D_FF, D_MODEL);

  // 8) FFN down + residual -> final output (f32)
  k_gemm_bf16<0, 1, 0><<<gqkv, 256, 0, stream>>>(h_bf, w2_bf, b2, x1,
      out_x, nullptr, ROWS, D_MODEL, D_FF);
}
